// TempCatStatic_86079734546560
// MI455X (gfx1250) — compile-verified, matrix operand order checked by reading, not run on hardware
//
#include <hip/hip_runtime.h>

// ---------------------------------------------------------------------------
// Types for CDNA5 WMMA (wave32): v_wmma_f32_16x16x32_bf16
// ---------------------------------------------------------------------------
typedef __bf16  v16bf __attribute__((ext_vector_type(16)));
typedef float   v8f   __attribute__((ext_vector_type(8)));

union FragU { unsigned u[8]; v16bf v; };

#define Bsz   8192
#define Tsz   24
#define Hsz   256
#define Ssz   256
#define Esz   256
#define BHsz  (Bsz * Hsz)

// f32 -> bf16 round-to-nearest-even
__device__ __forceinline__ unsigned short f32_to_bf16(float x) {
    unsigned u = __builtin_bit_cast(unsigned, x);
    u += 0x7fffu + ((u >> 16) & 1u);
    return (unsigned short)(u >> 16);
}

// Load a 16x32 (rows x K) bf16 fragment from a row-major [*, ld] bf16 matrix.
// ISA 7.12.2 16-bit A layout: lanes 0-15 -> row, lanes 16-31 same rows other
// K-half; VGPR v holds contiguous K-pair:
//   v in 0..3 : k0 = 2v + 8*(lane>=16)     (bytes [0..16) / [16..32))
//   v in 4..7 : k0 = 16 + 2(v-4) + 8*kh    (bytes [32..48) / [48..64))
// Per lane-half the 8 dwords are two contiguous 16B runs -> compiler merges
// them into b128 loads (global or ds, depending on the pointer's space).
// The same mapping with lane->n is the B operand of (X @ W^T), since
// B[k][n] = W[n][k] and each K-pair is contiguous in row-major W.
__device__ __forceinline__ v16bf load_frag16x32(const unsigned short* __restrict__ base,
                                                int row0, int ld, int kbase, int lane) {
    const int r  = row0 + (lane & 15);
    const int kh = (lane >> 4) & 1;
    const unsigned short* p = base + r * ld + kbase;
    FragU f;
#pragma unroll
    for (int v = 0; v < 8; ++v) {
        const int k0 = (v < 4) ? (2 * v + 8 * kh) : (16 + 2 * (v - 4) + 8 * kh);
        f.u[v] = *reinterpret_cast<const unsigned*>(p + k0);
    }
    return f.v;
}

__device__ __forceinline__ float fast_sigmoid(float x) {
    return 1.0f / (1.0f + __expf(-x));
}
__device__ __forceinline__ float fast_tanh(float x) {
    return 2.0f / (1.0f + __expf(-2.0f * x)) - 1.0f;
}

// ---------------------------------------------------------------------------
// Prep kernels
// ---------------------------------------------------------------------------
__global__ void cvt_f32_bf16_kernel(const float* __restrict__ in,
                                    unsigned short* __restrict__ out, int n) {
    int i = blockIdx.x * blockDim.x + threadIdx.x;
    if (i < n) out[i] = f32_to_bf16(in[i]);
}

__global__ void bias_fuse_kernel(const float* __restrict__ b_ih,
                                 const float* __restrict__ b_hh,
                                 float* __restrict__ bsum, int n) {
    int i = blockIdx.x * blockDim.x + threadIdx.x;
    if (i < n) bsum[i] = b_ih[i] + b_hh[i];
}

// ---------------------------------------------------------------------------
// Fused LSTM step. Block = 256 thr (8 waves) x 128 batch rows x 16 hidden
// cols. The block's W_hh slice (4 gates x 16 rows x 256 K bf16 = 32 KB) is
// staged into LDS once, so each wave's B fragments come from ds_load_b128
// instead of 8x-redundant L2 reads. A (h) streams from global as b128.
// ---------------------------------------------------------------------------
__global__ __launch_bounds__(256) void lstm_step_kernel(
    const unsigned short* __restrict__ h_in,    // [B][H] bf16 (prev step)
    unsigned short* __restrict__ h_out_bf16,    // [B][H] bf16 (next step)
    float* __restrict__ h_out_f32,              // [B][H]
    float* __restrict__ c_state,                // [B][H] in/out
    const unsigned short* __restrict__ Whh,     // [4H][H] bf16 row-major
    const float* __restrict__ W_ih,             // [4H][2] f32 row-major
    const float* __restrict__ bsum,             // [4H]  (b_ih + b_hh)
    const float* __restrict__ tfeat,            // [B][T]
    const float* __restrict__ tmask,            // [B][T]
    int t)
{
    __shared__ unsigned short Wlds[64 * Hsz];   // [4*16][256] bf16 = 32 KB

    const int tid  = threadIdx.x;
    const int lane = tid & 31;
    const int wave = tid >> 5;                     // 0..7 : M tile in block
    const int m0   = blockIdx.x * 128 + wave * 16; // batch row base
    const int j0   = blockIdx.y * 16;              // hidden col base
    const int kh   = lane >> 4;
    const int col  = j0 + (lane & 15);             // this lane's N column

    // Cooperative stage of W_hh slice: 64 rows x 512 B = 2048 uint4.
#pragma unroll
    for (int i = 0; i < 8; ++i) {
        const int u4   = i * 256 + tid;        // uint4 index, 0..2047
        const int row  = u4 >> 5;              // 32 uint4 per 512B row
        const int c4   = u4 & 31;
        const int gate = row >> 4;
        const int rr   = row & 15;
        const uint4 v = *reinterpret_cast<const uint4*>(
            Whh + (gate * Hsz + j0 + rr) * Hsz + c4 * 8);
        *reinterpret_cast<uint4*>(Wlds + row * Hsz + c4 * 8) = v;
    }
    __syncthreads();

    v8f acc[4];
#pragma unroll
    for (int g = 0; g < 4; ++g)
        acc[g] = (v8f){0.f, 0.f, 0.f, 0.f, 0.f, 0.f, 0.f, 0.f};

#pragma unroll
    for (int kc = 0; kc < Hsz; kc += 32) {
        v16bf a = load_frag16x32(h_in, m0, Hsz, kc, lane);
#pragma unroll
        for (int g = 0; g < 4; ++g) {
            v16bf b = load_frag16x32(Wlds, g * 16, Hsz, kc, lane);
            acc[g] = __builtin_amdgcn_wmma_f32_16x16x32_bf16(
                false, a, false, b, (short)0, acc[g], false, false);
        }
    }

    // Per-lane x-path coefficients for this column (hoisted out of r loop)
    float w0[4], w1[4], bb[4];
#pragma unroll
    for (int g = 0; g < 4; ++g) {
        const int gc = g * Hsz + col;
        w0[g] = W_ih[gc * 2 + 0];
        w1[g] = W_ih[gc * 2 + 1];
        bb[g] = bsum[gc];
    }

    // D layout (32-bit C/D 16x16): VGPR r -> M = r (+8 for lanes 16..31)
#pragma unroll
    for (int r = 0; r < 8; ++r) {
        const int b   = m0 + r + 8 * kh;
        const float x0 = tfeat[b * Tsz + t];
        const float x1 = tmask[b * Tsz + t];

        const float gi = acc[0][r] + x0 * w0[0] + x1 * w1[0] + bb[0];
        const float gf = acc[1][r] + x0 * w0[1] + x1 * w1[1] + bb[1];
        const float gg = acc[2][r] + x0 * w0[2] + x1 * w1[2] + bb[2];
        const float go = acc[3][r] + x0 * w0[3] + x1 * w1[3] + bb[3];

        const float i_ = fast_sigmoid(gi);
        const float f_ = fast_sigmoid(gf);
        const float g_ = fast_tanh(gg);
        const float o_ = fast_sigmoid(go);

        const int idx = b * Hsz + col;
        const float cn = f_ * c_state[idx] + i_ * g_;
        c_state[idx] = cn;
        const float hn = o_ * fast_tanh(cn);
        h_out_f32[idx]  = hn;
        h_out_bf16[idx] = f32_to_bf16(hn);
    }
}

// ---------------------------------------------------------------------------
// Fused MLP layer: out = relu(A @ W^T + bias), bf16 WMMA, f32 accumulate.
// W slice (16 rows x 256 K = 8 KB) staged in LDS, shared by the 8 waves.
// ---------------------------------------------------------------------------
__global__ __launch_bounds__(256) void mlp_layer_kernel(
    const unsigned short* __restrict__ A,     // [B][256] bf16
    const unsigned short* __restrict__ W,     // [256][256] bf16 row-major
    const float* __restrict__ bias,           // [256]
    unsigned short* __restrict__ out_bf16,    // [B][256]
    float* __restrict__ out_f32)              // [B][256]
{
    __shared__ unsigned short Wlds[16 * Esz]; // 8 KB

    const int tid  = threadIdx.x;
    const int lane = tid & 31;
    const int wave = tid >> 5;
    const int m0   = blockIdx.x * 128 + wave * 16;
    const int j0   = blockIdx.y * 16;
    const int kh   = lane >> 4;
    const int col  = j0 + (lane & 15);

    // Stage 16 rows x 512 B = 512 uint4 with 256 threads (2 each).
#pragma unroll
    for (int i = 0; i < 2; ++i) {
        const int u4  = i * 256 + tid;       // 0..511
        const int row = u4 >> 5;
        const int c4  = u4 & 31;
        const uint4 v = *reinterpret_cast<const uint4*>(
            W + (j0 + row) * Esz + c4 * 8);
        *reinterpret_cast<uint4*>(Wlds + row * Esz + c4 * 8) = v;
    }
    __syncthreads();

    v8f acc = (v8f){0.f, 0.f, 0.f, 0.f, 0.f, 0.f, 0.f, 0.f};
#pragma unroll
    for (int kc = 0; kc < Esz; kc += 32) {
        v16bf a = load_frag16x32(A, m0, Esz, kc, lane);
        v16bf b = load_frag16x32(Wlds, 0, Esz, kc, lane);
        acc = __builtin_amdgcn_wmma_f32_16x16x32_bf16(
            false, a, false, b, (short)0, acc, false, false);
    }

    const float bv = bias[col];
#pragma unroll
    for (int r = 0; r < 8; ++r) {
        const int b   = m0 + r + 8 * kh;
        const int idx = b * Esz + col;
        float v = acc[r] + bv;
        v = fmaxf(v, 0.0f);
        out_f32[idx]  = v;
        out_bf16[idx] = f32_to_bf16(v);
    }
}

// ---------------------------------------------------------------------------
// Head: predicts = [h, s] @ W_lr^T + b_lr ; BCE-with-logits partial sums.
// One thread per batch row, float4-vectorized dot; block-reduced loss.
// ---------------------------------------------------------------------------
__global__ __launch_bounds__(256) void head_kernel(
    const float* __restrict__ h,        // [B][H]
    const float* __restrict__ s,        // [B][E]
    const float* __restrict__ Wlr,      // [1][H+E]
    const float* __restrict__ blr,      // [1]
    const float* __restrict__ targets,  // [B][1]
    float* __restrict__ predicts,       // [B]
    float* __restrict__ loss_acc)       // [1]
{
    const int b = blockIdx.x * blockDim.x + threadIdx.x;

    float p = blr[0];
    const float4* hr = reinterpret_cast<const float4*>(h + b * Hsz);
    const float4* sr = reinterpret_cast<const float4*>(s + b * Esz);
    const float4* w0 = reinterpret_cast<const float4*>(Wlr);
    const float4* w1 = reinterpret_cast<const float4*>(Wlr + Hsz);
#pragma unroll 4
    for (int j = 0; j < Hsz / 4; ++j) {
        const float4 a = hr[j], w = w0[j];
        p += a.x * w.x + a.y * w.y + a.z * w.z + a.w * w.w;
    }
#pragma unroll 4
    for (int j = 0; j < Esz / 4; ++j) {
        const float4 a = sr[j], w = w1[j];
        p += a.x * w.x + a.y * w.y + a.z * w.z + a.w * w.w;
    }

    predicts[b] = p;

    const float tg = targets[b];
    const float l  = fmaxf(p, 0.0f) - p * tg + log1pf(__expf(-fabsf(p)));

    __shared__ float red[256];
    red[threadIdx.x] = l;
    __syncthreads();
#pragma unroll
    for (int st = 128; st > 0; st >>= 1) {
        if (threadIdx.x < st) red[threadIdx.x] += red[threadIdx.x + st];
        __syncthreads();
    }
    if (threadIdx.x == 0) atomicAdd(loss_acc, red[0]);
}

__global__ void finalize_loss_kernel(const float* __restrict__ loss_acc,
                                     float* __restrict__ out) {
    out[0] = loss_acc[0] * (1.0f / (float)Bsz);
}

// ---------------------------------------------------------------------------
// Host launcher
// ---------------------------------------------------------------------------
extern "C" void kernel_launch(void* const* d_in, const int* in_sizes, int n_in,
                              void* d_out, int out_size, void* d_ws, size_t ws_size,
                              hipStream_t stream) {
    (void)in_sizes; (void)n_in; (void)out_size; (void)ws_size;

    const float* static_f = (const float*)d_in[0];   // [B,S]
    const float* tfeat    = (const float*)d_in[1];   // [B,T]
    const float* tmask    = (const float*)d_in[2];   // [B,T]
    const float* targets  = (const float*)d_in[3];   // [B,1]
    const float* h0       = (const float*)d_in[4];   // [B,H]
    const float* c0       = (const float*)d_in[5];   // [B,H]
    const float* W_ih     = (const float*)d_in[6];   // [4H,2]
    const float* W_hh     = (const float*)d_in[7];   // [4H,H]
    const float* b_ih     = (const float*)d_in[8];   // [4H]
    const float* b_hh     = (const float*)d_in[9];   // [4H]
    const float* W1       = (const float*)d_in[10];  // [E,S]
    const float* b1       = (const float*)d_in[11];
    const float* W2       = (const float*)d_in[12];
    const float* b2       = (const float*)d_in[13];
    const float* W3       = (const float*)d_in[14];
    const float* b3       = (const float*)d_in[15];
    const float* W_lr     = (const float*)d_in[16];  // [1, H+E]
    const float* b_lr     = (const float*)d_in[17];

    // ---- workspace layout ----
    size_t off = 0;
    auto take = [&](size_t bytes) -> char* {
        char* p = (char*)d_ws + off;
        off += (bytes + 255) & ~(size_t)255;
        return p;
    };
    unsigned short* hA   = (unsigned short*)take((size_t)BHsz * 2); // bf16 h ping
    unsigned short* hB   = (unsigned short*)take((size_t)BHsz * 2); // bf16 h pong
    float*          hF   = (float*)take((size_t)BHsz * 4);          // f32 h
    float*          cW   = (float*)take((size_t)BHsz * 4);          // f32 c
    unsigned short* sA   = (unsigned short*)take((size_t)BHsz * 2); // bf16 s ping
    unsigned short* sB   = (unsigned short*)take((size_t)BHsz * 2); // bf16 s pong
    float*          sF   = (float*)take((size_t)BHsz * 4);          // f32 s
    unsigned short* Whhb = (unsigned short*)take((size_t)4 * Hsz * Hsz * 2);
    unsigned short* W1b  = (unsigned short*)take((size_t)Esz * Ssz * 2);
    unsigned short* W2b  = (unsigned short*)take((size_t)Esz * Esz * 2);
    unsigned short* W3b  = (unsigned short*)take((size_t)Esz * Esz * 2);
    float*          bsum = (float*)take((size_t)4 * Hsz * 4);
    float*          loss = (float*)take(sizeof(float));

    // ---- prep: state copies, bf16 conversions, bias fuse ----
    hipMemsetAsync(loss, 0, sizeof(float), stream);
    hipMemcpyAsync(cW, c0, (size_t)BHsz * 4, hipMemcpyDeviceToDevice, stream);
    hipMemcpyAsync(hF, h0, (size_t)BHsz * 4, hipMemcpyDeviceToDevice, stream);

    auto cvt = [&](const float* in, unsigned short* out, int n) {
        cvt_f32_bf16_kernel<<<(n + 255) / 256, 256, 0, stream>>>(in, out, n);
    };
    cvt(h0,       hA,   BHsz);
    cvt(static_f, sA,   Bsz * Ssz);
    cvt(W_hh,     Whhb, 4 * Hsz * Hsz);
    cvt(W1,       W1b,  Esz * Ssz);
    cvt(W2,       W2b,  Esz * Esz);
    cvt(W3,       W3b,  Esz * Esz);
    bias_fuse_kernel<<<(4 * Hsz + 255) / 256, 256, 0, stream>>>(b_ih, b_hh, bsum, 4 * Hsz);

    // ---- 24 fused LSTM steps (sequential dependency) ----
    const dim3 grid(Bsz / 128, Hsz / 16);   // 64 x 16 blocks
    unsigned short* bufs[2] = {hA, hB};
    for (int t = 0; t < Tsz; ++t) {
        lstm_step_kernel<<<grid, 256, 0, stream>>>(
            bufs[t & 1], bufs[(t + 1) & 1], hF, cW,
            Whhb, W_ih, bsum, tfeat, tmask, t);
    }

    // ---- static MLP (3 fused layers, ping-pong bf16 activations) ----
    const dim3 gridS(Bsz / 128, Esz / 16);
    mlp_layer_kernel<<<gridS, 256, 0, stream>>>(sA, W1b, b1, sB, sF);
    mlp_layer_kernel<<<gridS, 256, 0, stream>>>(sB, W2b, b2, sA, sF);
    mlp_layer_kernel<<<gridS, 256, 0, stream>>>(sA, W3b, b3, sB, sF);

    // ---- classifier + BCE loss ----
    float* predicts = (float*)d_out;         // [0 .. B-1], then loss at [B]
    head_kernel<<<Bsz / 256, 256, 0, stream>>>(hF, sF, W_lr, b_lr, targets,
                                               predicts, loss);
    finalize_loss_kernel<<<1, 1, 0, stream>>>(loss, predicts + Bsz);
}
